// CrossAttention_72877005078974
// MI455X (gfx1250) — matrix operand order repaired and checked by reading, not compiled, not run on hardware
//
#include <hip/hip_runtime.h>
#include <hip/hip_bf16.h>

typedef __attribute__((ext_vector_type(16))) _Float16 v16h;
typedef __attribute__((ext_vector_type(8)))  _Float16 v8h;
typedef __attribute__((ext_vector_type(4)))  _Float16 v4h;
typedef __attribute__((ext_vector_type(8)))  float    v8f;
typedef __attribute__((ext_vector_type(4)))  int      v4i;

typedef __attribute__((address_space(1))) v4i as1_v4i;
typedef __attribute__((address_space(3))) v4i as3_v4i;

#define B_     4
#define N_     4096
#define L_     256
#define QD_    512
#define CD_    768
#define INNER_ 512
#define DHEAD_ 64

#if defined(__gfx1250__) && __has_builtin(__builtin_amdgcn_global_load_async_to_lds_b128)
#define HAVE_ASYNC_LDS 1
#else
#define HAVE_ASYNC_LDS 0
#endif

__device__ __forceinline__ void async_wait0() {
#if defined(__gfx1250__)
#if __has_builtin(__builtin_amdgcn_s_wait_asynccnt)
  __builtin_amdgcn_s_wait_asynccnt(0);
#elif HAVE_ASYNC_LDS
  asm volatile("s_wait_asynccnt 0x0" ::: "memory");
#endif
#endif
}

// Cooperatively stage a 32KB (L_*DHEAD_ halves) tile global -> LDS with 128 threads.
// Async path: GLOBAL_LOAD_ASYNC_TO_LDS_B128 (ASYNCcnt), else load+ds_store.
__device__ __forceinline__ void stage_32kb(const _Float16* __restrict__ g,
                                           _Float16* l, int tid) {
#pragma unroll
  for (int c = 0; c < 16; ++c) {
    int off = (c * 128 + tid) * 8;            // halves; 16B per thread per step
#if HAVE_ASYNC_LDS
    __builtin_amdgcn_global_load_async_to_lds_b128(
        (as1_v4i*)(uintptr_t)(g + off),
        (as3_v4i*)(unsigned)(uintptr_t)(l + off), 0, 0);
#else
    *(v8h*)(l + off) = *(const v8h*)(g + off);
#endif
  }
}

// ---------------- WMMA fragment helpers (gfx1250 wave32 layouts) ----------------
__device__ __forceinline__ v16h frag_a_f16(const _Float16* p, int ld, int lane) {
  const _Float16* base = p + (size_t)(lane & 15) * ld + ((lane >> 4) << 3);
  v8h lo = *(const v8h*)(base);
  v8h hi = *(const v8h*)(base + 16);
  v16h r;
#pragma unroll
  for (int i = 0; i < 8; ++i) { r[i] = lo[i]; r[i + 8] = hi[i]; }
  return r;
}

__device__ __forceinline__ v16h frag_b_f16(const _Float16* p, int ld, int lane) {
  const _Float16* base = p + (size_t)(lane & 15) * ld + ((lane >> 4) << 3);
  v8h lo = *(const v8h*)(base);
  v8h hi = *(const v8h*)(base + 16);
  v16h r;
#pragma unroll
  for (int i = 0; i < 8; ++i) { r[i] = lo[i]; r[i + 8] = hi[i]; }
  return r;
}

__device__ __forceinline__ v8f wmma_f16(v16h a, v16h b, v8f c) {
  return __builtin_amdgcn_wmma_f32_16x16x32_f16(false, a, false, b, (short)0, c,
                                                false, false);
}

// ---------------- k_wt: weight transpose + f32 -> f16 ----------------
__global__ void k_wt(const float* __restrict__ src, _Float16* __restrict__ dst,
                     int rows, int cols) {
  int i = blockIdx.x * blockDim.x + threadIdx.x;
  if (i >= rows * cols) return;
  int r = i / cols, c = i - r * cols;
  dst[(size_t)c * rows + r] = (_Float16)src[i];
}

// ---------------- k_cvt: bulk f32 -> f16 ----------------
__global__ void k_cvt(const float* __restrict__ src, _Float16* __restrict__ dst,
                      int n4) {
  int i = blockIdx.x * blockDim.x + threadIdx.x;
  if (i >= n4) return;
  float4 v = ((const float4*)src)[i];
  v4h o = {(_Float16)v.x, (_Float16)v.y, (_Float16)v.z, (_Float16)v.w};
  ((v4h*)dst)[i] = o;
}

// ---------------- k_mask_pack: coalesced gather -> L2-resident bitmask ---------
// mask_bool(b,n,l) = mask[b][l][2*(n>>6)][2*(n&63)] != 0
// mpack[b][l]: 128 dwords, bit (n&31) of dword (n>>5).
__global__ void k_mask_pack(const int* __restrict__ mask,
                            unsigned* __restrict__ mpack) {
  int idx = blockIdx.x * blockDim.x + threadIdx.x;      // B*L*64
  if (idx >= B_ * L_ * 64) return;
  int i = idx & 63;
  int l = (idx >> 6) & (L_ - 1);
  int b = idx >> 14;
  const int* row = mask + ((size_t)(b * L_ + l) * 128 + 2 * i) * 128;
  unsigned w0 = 0, w1 = 0;
#pragma unroll
  for (int j = 0; j < 32; ++j) if (row[2 * j])        w0 |= 1u << j;
#pragma unroll
  for (int j = 0; j < 32; ++j) if (row[2 * (j + 32)]) w1 |= 1u << j;
  unsigned* o = mpack + (size_t)(b * L_ + l) * 128 + 2 * i;
  o[0] = w0;
  o[1] = w1;
}

// ---------------- k_proj_kv: K and V^T projections ----------------
__global__ void k_proj_kv(const _Float16* __restrict__ ctxh,   // [B][L][CD]
                          const _Float16* __restrict__ WkT,    // [INNER][CD]
                          const _Float16* __restrict__ WvT,    // [INNER][CD]
                          _Float16* __restrict__ k_mat,        // [32][L][64]
                          _Float16* __restrict__ vT_mat) {     // [32][64][L]
  int wid  = blockIdx.x * (blockDim.x >> 5) + (threadIdx.x >> 5);
  int lane = threadIdx.x & 31;
  int which = wid >> 11;
  int t  = wid & 2047;
  int bh = t >> 6;
  int lt = (t >> 2) & 15;
  int dt = t & 3;
  int b = bh >> 3, h = bh & 7;
  const _Float16* WT = which ? WvT : WkT;
  const _Float16* A0 = ctxh + (size_t)b * L_ * CD_ + (size_t)(lt * 16) * CD_;
  const _Float16* B0 = WT + (size_t)(h * DHEAD_ + dt * 16) * CD_;
  v8f c = {};
#pragma unroll 4
  for (int cc = 0; cc < CD_; cc += 32)
    c = wmma_f16(frag_a_f16(A0 + cc, CD_, lane), frag_b_f16(B0 + cc, CD_, lane), c);
  int col = lane & 15, hf = lane >> 4;
  if (which == 0) {
    _Float16* o = k_mat + (size_t)bh * L_ * DHEAD_ + (size_t)(lt * 16) * DHEAD_ +
                  dt * 16 + col;
#pragma unroll
    for (int r = 0; r < 8; ++r) o[(size_t)(8 * hf + r) * DHEAD_] = (_Float16)c[r];
  } else {
    _Float16* o = vT_mat + (size_t)bh * DHEAD_ * L_ + (size_t)(dt * 16 + col) * L_ +
                  lt * 16 + 8 * hf;
#pragma unroll
    for (int r = 0; r < 8; ++r) o[r] = (_Float16)c[r];
  }
}

// ---------------- k_attn: fused Q-proj + sim + mask + softmax + attn@V ---------
// Block = 4 waves, all on one (b,h). K tile (32KB) async-staged to LDS and shared
// by the 4 waves; same buffer is re-staged with V^T (overlapped with softmax).
__global__ void __launch_bounds__(128) k_attn(
    const _Float16* __restrict__ xh,        // [B][N][QD]
    const unsigned* __restrict__ mpack,     // [B][L][128]
    const _Float16* __restrict__ WqT,       // [INNER][QD]
    const _Float16* __restrict__ k_mat,     // [32][L][64]
    const _Float16* __restrict__ vT_mat,    // [32][64][L]
    _Float16* __restrict__ attn_out) {      // [B][N][INNER]
  __shared__ __align__(32) _Float16 s_kv[L_ * DHEAD_];     // 32KB: K, then V^T
  __shared__ __align__(32) _Float16 s_t[4][16 * DHEAD_];   // 2KB / wave transpose
  int warp = threadIdx.x >> 5, lane = threadIdx.x & 31;
  int bh = blockIdx.x >> 6;                 // 32
  int nt = (blockIdx.x & 63) * 4 + warp;    // 256 n-tiles
  int b = bh >> 3, h = bh & 7;
  int n0 = nt * 16;
  int col = lane & 15, hf = lane >> 4;
  _Float16* myt = &s_t[warp][0];
  const float scale = 0.125f;               // 64^-0.5

  // issue async K staging; overlap with Q projection below
  stage_32kb(k_mat + (size_t)bh * L_ * DHEAD_, s_kv, threadIdx.x);

  // ---- Q tile = xh @ Wq[:, head]  (16x64, K=512) ----
  v8f cq[4] = {};
  const _Float16* A0 = xh + (size_t)b * N_ * QD_ + (size_t)n0 * QD_;
#pragma unroll 1
  for (int cc = 0; cc < QD_; cc += 32) {
    __builtin_prefetch(A0 + cc + 128, 0, 1);
    v16h a = frag_a_f16(A0 + cc, QD_, lane);
#pragma unroll
    for (int dt = 0; dt < 4; ++dt) {
      v16h bf = frag_b_f16(WqT + (size_t)(h * DHEAD_ + dt * 16) * QD_ + cc, QD_, lane);
      cq[dt] = wmma_f16(a, bf, cq[dt]);
    }
  }
  // q * scale -> per-wave LDS [16][64], reload as A fragments
#pragma unroll
  for (int dt = 0; dt < 4; ++dt)
#pragma unroll
    for (int r = 0; r < 8; ++r)
      myt[(8 * hf + r) * DHEAD_ + dt * 16 + col] = (_Float16)(cq[dt][r] * scale);
  v16h qa0 = frag_a_f16(myt + 0,  DHEAD_, lane);
  v16h qa1 = frag_a_f16(myt + 32, DHEAD_, lane);

  // ---- mask bits: all 16 q-rows of this wave live in one dword per l ----
  const unsigned* mp = mpack + (size_t)b * L_ * 128 + (n0 >> 5);
  unsigned mw[16];
#pragma unroll
  for (int t = 0; t < 16; ++t)
    mw[t] = mp[(size_t)(t * 16 + col) * 128];

  async_wait0();
  __syncthreads();                          // K tile visible to all 4 waves

  // ---- sim = q @ K^T from LDS ----
  v8f cs[16];
#pragma unroll
  for (int t = 0; t < 16; ++t) {
    const _Float16* K0 = s_kv + t * 16 * DHEAD_;
    v8f c = {};
    c = wmma_f16(qa0, frag_b_f16(K0 + 0,  DHEAD_, lane), c);
    c = wmma_f16(qa1, frag_b_f16(K0 + 32, DHEAD_, lane), c);
    cs[t] = c;
  }

  __syncthreads();                          // everyone done reading K
  // issue async V^T staging into the same buffer; overlap with softmax (regs only)
  stage_32kb(vT_mat + (size_t)bh * DHEAD_ * L_, s_kv, threadIdx.x);

  // ---- mask + softmax over l; normalized attn kept in cs registers ----
#pragma unroll
  for (int r = 0; r < 8; ++r) {
    int bitpos = (n0 & 31) + 8 * hf + r;
    float vreg[16];
    float mx = -3.0e38f;
#pragma unroll
    for (int t = 0; t < 16; ++t) {
      float s = cs[t][r] + (((mw[t] >> bitpos) & 1u) ? 0.0f : -1.0e9f);
      vreg[t] = s;
      mx = fmaxf(mx, s);
    }
#pragma unroll
    for (int off = 1; off < 16; off <<= 1) mx = fmaxf(mx, __shfl_xor(mx, off, 32));
    float sum = 0.0f;
#pragma unroll
    for (int t = 0; t < 16; ++t) { float e = __expf(vreg[t] - mx); vreg[t] = e; sum += e; }
#pragma unroll
    for (int off = 1; off < 16; off <<= 1) sum += __shfl_xor(sum, off, 32);
    float inv = 1.0f / sum;
#pragma unroll
    for (int t = 0; t < 16; ++t) cs[t][r] = vreg[t] * inv;
  }

  async_wait0();
  __syncthreads();                          // V^T tile visible

  // ---- out = attn @ V : chunked 16x32 transposes through per-wave LDS ----
  v8f co[4] = {};
#pragma unroll
  for (int ks = 0; ks < 8; ++ks) {
#pragma unroll
    for (int u = 0; u < 2; ++u)
#pragma unroll
      for (int r = 0; r < 8; ++r)
        myt[(8 * hf + r) * 32 + u * 16 + col] = (_Float16)cs[2 * ks + u][r];
    v16h a = frag_a_f16(myt, 32, lane);
    const _Float16* V0 = s_kv + ks * 32;
#pragma unroll
    for (int dt = 0; dt < 4; ++dt)
      co[dt] = wmma_f16(a, frag_b_f16(V0 + (size_t)(dt * 16) * L_, L_, lane), co[dt]);
  }
#pragma unroll
  for (int dt = 0; dt < 4; ++dt) {
    _Float16* o = attn_out + ((size_t)(b * N_ + n0 + 8 * hf)) * INNER_ +
                  h * DHEAD_ + dt * 16 + col;
#pragma unroll
    for (int r = 0; r < 8; ++r) o[(size_t)r * INNER_] = (_Float16)co[dt][r];
  }
}

// ---------------- k_outproj: out = attn_out @ Wo + bo, 4 col-tiles per wave ----
__global__ void k_outproj(const _Float16* __restrict__ attn_out,
                          const _Float16* __restrict__ WoT,
                          const float* __restrict__ bo,
                          float* __restrict__ out) {
  int wid  = blockIdx.x * (blockDim.x >> 5) + (threadIdx.x >> 5);
  int lane = threadIdx.x & 31;
  int mt = wid >> 3;
  int ng = wid & 7;
  int col = lane & 15, hf = lane >> 4;
  const _Float16* A0 = attn_out + (size_t)mt * 16 * INNER_;
  const _Float16* B0 = WoT + (size_t)(ng * 64) * INNER_;
  v8f c[4] = {};
#pragma unroll 1
  for (int k = 0; k < INNER_; k += 32) {
    __builtin_prefetch(A0 + k + 128, 0, 1);
    v16h a = frag_a_f16(A0 + k, INNER_, lane);
#pragma unroll
    for (int j = 0; j < 4; ++j)
      c[j] = wmma_f16(a, frag_b_f16(B0 + (size_t)(j * 16) * INNER_ + k, INNER_, lane),
                      c[j]);
  }
#pragma unroll
  for (int j = 0; j < 4; ++j) {
    float bias = bo[ng * 64 + j * 16 + col];
    float* o = out + ((size_t)(mt * 16) + 8 * hf) * QD_ + ng * 64 + j * 16 + col;
#pragma unroll
    for (int r = 0; r < 8; ++r) o[(size_t)r * QD_] = c[j][r] + bias;
  }
}

// ---------------- launch ----------------
extern "C" void kernel_launch(void* const* d_in, const int* in_sizes, int n_in,
                              void* d_out, int out_size, void* d_ws, size_t ws_size,
                              hipStream_t stream) {
  const float* x       = (const float*)d_in[0];
  const float* context = (const float*)d_in[1];
  const int*   mask    = (const int*)d_in[2];
  const float* Wq      = (const float*)d_in[3];
  const float* Wk      = (const float*)d_in[4];
  const float* Wv      = (const float*)d_in[5];
  const float* Wo      = (const float*)d_in[6];
  const float* bo      = (const float*)d_in[7];
  float* out = (float*)d_out;

  char* ws = (char*)d_ws;
  _Float16* WqT  = (_Float16*)(ws + 0);
  _Float16* WkT  = (_Float16*)(ws + 524288);
  _Float16* WvT  = (_Float16*)(ws + 1310720);
  _Float16* WoT  = (_Float16*)(ws + 2097152);
  _Float16* kmat = (_Float16*)(ws + 2621440);
  _Float16* vT   = (_Float16*)(ws + 3670016);
  _Float16* aout = (_Float16*)(ws + 4718592);
  _Float16* xh   = (_Float16*)(ws + 21495808);
  _Float16* ctxh = (_Float16*)(ws + 38273024);
  unsigned* mpk  = (unsigned*)(ws + 39845888);   // total ~38.5 MB

  k_wt<<<(512 * 512 + 255) / 256, 256, 0, stream>>>(Wq, WqT, 512, 512);
  k_wt<<<(768 * 512 + 255) / 256, 256, 0, stream>>>(Wk, WkT, 768, 512);
  k_wt<<<(768 * 512 + 255) / 256, 256, 0, stream>>>(Wv, WvT, 768, 512);
  k_wt<<<(512 * 512 + 255) / 256, 256, 0, stream>>>(Wo, WoT, 512, 512);
  k_cvt<<<(B_ * N_ * QD_ / 4 + 255) / 256, 256, 0, stream>>>(x, xh, B_ * N_ * QD_ / 4);
  k_cvt<<<(B_ * L_ * CD_ / 4 + 255) / 256, 256, 0, stream>>>(context, ctxh,
                                                             B_ * L_ * CD_ / 4);
  k_mask_pack<<<(B_ * L_ * 64 + 255) / 256, 256, 0, stream>>>(mask, mpk);
  k_proj_kv<<<512, 256, 0, stream>>>(ctxh, WkT, WvT, kmat, vT);
  k_attn<<<2048, 128, 0, stream>>>(xh, mpk, WqT, kmat, vT, aout);
  k_outproj<<<1024, 256, 0, stream>>>(aout, WoT, bo, out);
}
